// AttnDecoderRNN_6313601925654
// MI455X (gfx1250) — matrix operand order repaired and checked by reading, not compiled, not run on hardware
//
#include <hip/hip_runtime.h>

// ---------------------------------------------------------------------------
// AttnDecoderRNN for MI455X (gfx1250): bf16 WMMA GEMMs + fused attention.
// Weights converted to bf16 once per launch -> entire weight set (~110MB)
// is L2-resident (192MB L2), all 64 steps stream from L2 instead of HBM.
// Matrix math uses v_wmma_f32_16x16x32_bf16 with f32 accumulation.
// Each GEMM wave owns a 32x16 C slab: two WMMAs share one B (weight)
// fragment, halving weight traffic (the dominant data movement).
// ---------------------------------------------------------------------------

constexpr int NB  = 32;          // batch
constexpr int NT  = 64;          // decode steps
constexpr int NTE = 128;         // encoder timesteps
constexpr int NH  = 1024;        // hidden
constexpr int NE  = 512;         // embedding
constexpr int NV  = 32000;       // vocab
constexpr int NKC = NE + NH + NH;  // 2560 = concat K for gates GEMM
constexpr int NG  = 4 * NH;        // 4096 gates
constexpr int SOS_TOK = 1;

typedef __attribute__((ext_vector_type(16))) __bf16       v16bf;
typedef __attribute__((ext_vector_type(8)))  float        v8f;
typedef __attribute__((ext_vector_type(4)))  unsigned int v4u;

#define DEV_INLINE __device__ __forceinline__

DEV_INLINE float sigmoidf(float x) { return 1.f / (1.f + expf(-x)); }

// ---------------------------------------------------------------------------
// WMMA GEMM:  C[M,N] = A[M,K] * W[N,K]^T   (bf16 in, f32 out), M % 32 == 0.
// One wave per 32x16 C slab (two 16x16 WMMA tiles sharing the B fragment).
// Fragment layouts used here (contiguous k in A and B; ISA table order differs):
//  A 16x32 bf16 : lane = M row (0-15) + half(lane>>4); per lane the 16 bf16
//                 are K = {half*16 .. +15} (16 consecutive) -> 2x b128.
//  B 32x16 bf16 : lane = N col (0-15) + half; per lane K = {half*16 .. +15}
//                 (16 consecutive) -> 2x b128 from row-major W.
//  D 16x16 f32  : lane = N col; VGPR r -> M = r + (lane>>4)*8.
// ---------------------------------------------------------------------------
__global__ __launch_bounds__(256)
void wmma_gemm_nt(float* __restrict__ C, const __bf16* __restrict__ A,
                  const __bf16* __restrict__ W,
                  int Mpairs, int Ntiles, int K, int lda, int ldw, int ldc)
{
    int wid  = (int)((blockIdx.x * blockDim.x + threadIdx.x) >> 5);
    int lane = threadIdx.x & 31;
    if (wid >= Mpairs * Ntiles) return;       // wave-uniform: EXEC stays all-1s
    int nt = wid % Ntiles;
    int mp = wid / Ntiles;
    int mbase = mp * 32, nbase = nt * 16;
    int half  = lane >> 4;

    const __bf16* ap0 = A + (size_t)(mbase + (lane & 15)) * lda + half * 16;
    const __bf16* ap1 = ap0 + (size_t)16 * lda;
    const __bf16* bp  = W + (size_t)(nbase + (lane & 15)) * ldw + half * 16;

    v8f acc0 = {0.f, 0.f, 0.f, 0.f, 0.f, 0.f, 0.f, 0.f};
    v8f acc1 = {0.f, 0.f, 0.f, 0.f, 0.f, 0.f, 0.f, 0.f};
    for (int k = 0; k < K; k += 32) {
        union { v4u u[2]; v16bf v; } ua0, ua1, ub;
        ub.u[0]  = *(const v4u*)(bp + k);        // K = k+half*16   .. +7
        ub.u[1]  = *(const v4u*)(bp + k + 8);    // K = k+half*16+8 .. +15
        ua0.u[0] = *(const v4u*)(ap0 + k);       // K = k+half*16   .. +7
        ua0.u[1] = *(const v4u*)(ap0 + k + 8);   // K = k+half*16+8 .. +15
        ua1.u[0] = *(const v4u*)(ap1 + k);
        ua1.u[1] = *(const v4u*)(ap1 + k + 8);
        acc0 = __builtin_amdgcn_wmma_f32_16x16x32_bf16(
                   false, ua0.v, false, ub.v, (short)0, acc0, false, false);
        acc1 = __builtin_amdgcn_wmma_f32_16x16x32_bf16(
                   false, ua1.v, false, ub.v, (short)0, acc1, false, false);
    }

    float* cp0 = C + (size_t)(mbase + half * 8) * ldc + nbase + (lane & 15);
    float* cp1 = cp0 + (size_t)16 * ldc;
    #pragma unroll
    for (int r = 0; r < 8; ++r) cp0[(size_t)r * ldc] = acc0[r];
    #pragma unroll
    for (int r = 0; r < 8; ++r) cp1[(size_t)r * ldc] = acc1[r];
}

// --------------------------- precompute kernels ----------------------------
__global__ void cvt_f32_bf16(__bf16* __restrict__ dst, const float* __restrict__ src, int n)
{
    for (int i = blockIdx.x * blockDim.x + threadIdx.x; i < n; i += gridDim.x * blockDim.x)
        dst[i] = (__bf16)src[i];
}

// W_cat[j, 0:1536] = W_ih[j,:] ; W_cat[j, 1536:2560] = W_hh[j,:]
__global__ void build_wcat(__bf16* __restrict__ wcat, const float* __restrict__ W_ih,
                           const float* __restrict__ W_hh)
{
    const int n = NG * NKC;
    for (int i = blockIdx.x * blockDim.x + threadIdx.x; i < n; i += gridDim.x * blockDim.x) {
        int row = i / NKC, col = i % NKC;
        float v = (col < NE + NH) ? W_ih[(size_t)row * (NE + NH) + col]
                                  : W_hh[(size_t)row * NH + (col - (NE + NH))];
        wcat[i] = (__bf16)v;
    }
}

// Teacher forcing: token(t,b) = (t==0 ? SOS : target[b,t-1]); padding_idx 0 -> zeros.
__global__ void build_emb(__bf16* __restrict__ emb_bf, const int* __restrict__ target,
                          const float* __restrict__ emb_tab)
{
    const int n = NT * NB * NE;
    for (int i = blockIdx.x * blockDim.x + threadIdx.x; i < n; i += gridDim.x * blockDim.x) {
        int t = i / (NB * NE), r = i % (NB * NE);
        int b = r / NE, e = r % NE;
        int tok = (t == 0) ? SOS_TOK : target[b * NT + (t - 1)];
        float v = (tok == 0) ? 0.f : emb_tab[(size_t)tok * NE + e];
        emb_bf[i] = (__bf16)v;
    }
}

__global__ void init_state(float* __restrict__ h, float* __restrict__ c,
                           __bf16* __restrict__ h_bf,
                           const float* __restrict__ eh, const float* __restrict__ ec)
{
    int gid = blockIdx.x * blockDim.x + threadIdx.x;
    if (gid < NB * NH) {
        float v = eh[gid];
        h[gid] = v;  h_bf[gid] = (__bf16)v;  c[gid] = ec[gid];
    }
}

// ---------------------------------------------------------------------------
// Fused Bahdanau attention: one workgroup (8 wave32) per batch row.
//   scores[t] = Va . tanh(q_b + encUa[b,t,:]) + Va_b ; softmax ; context.
// Also assembles the bf16 GEMM A-operand  a_bf[b] = [emb | context | h].
// ---------------------------------------------------------------------------
__global__ __launch_bounds__(256)
void attn_kernel(const float* __restrict__ q, const float* __restrict__ encUa,
                 const float* __restrict__ enc, const float* __restrict__ Va_w,
                 const float* __restrict__ Va_b, const __bf16* __restrict__ emb_t,
                 const __bf16* __restrict__ h_bf, __bf16* __restrict__ a_bf)
{
    __shared__ float s_q[NH], s_va[NH], s_w[NTE], s_red[256];
    __shared__ float s_max, s_sum;
    int b = blockIdx.x, tid = threadIdx.x;
    int wave = tid >> 5, lane = tid & 31;

    for (int i = tid; i < NH; i += 256) { s_q[i] = q[b * NH + i]; s_va[i] = Va_w[i]; }
    __syncthreads();

    // scores: each wave owns 16 encoder positions; lanes stride K (coalesced).
    float vab = Va_b[0];
    for (int t = wave * 16; t < wave * 16 + 16; ++t) {
        const float* eu = encUa + ((size_t)b * NTE + t) * NH;
        float p = 0.f;
        for (int k = lane; k < NH; k += 32)
            p += s_va[k] * tanhf(s_q[k] + eu[k]);
        for (int off = 16; off; off >>= 1) p += __shfl_xor(p, off, 32);
        if (lane == 0) s_w[t] = p + vab;
    }
    __syncthreads();

    // softmax over 128 positions
    s_red[tid] = (tid < NTE) ? s_w[tid] : -3.4e38f;
    __syncthreads();
    for (int s = 128; s; s >>= 1) { if (tid < s) s_red[tid] = fmaxf(s_red[tid], s_red[tid + s]); __syncthreads(); }
    if (tid == 0) s_max = s_red[0];
    __syncthreads();
    float e = 0.f;
    if (tid < NTE) { e = expf(s_w[tid] - s_max); s_w[tid] = e; }
    s_red[tid] = e;
    __syncthreads();
    for (int s = 128; s; s >>= 1) { if (tid < s) s_red[tid] += s_red[tid + s]; __syncthreads(); }
    if (tid == 0) s_sum = s_red[0];
    __syncthreads();

    // context: each thread owns 4 hidden dims (coalesced over encoder rows)
    float inv = 1.f / s_sum;
    float c0 = 0.f, c1 = 0.f, c2 = 0.f, c3 = 0.f;
    for (int t = 0; t < NTE; ++t) {
        float wt = s_w[t];
        const float* er = enc + ((size_t)b * NTE + t) * NH;
        c0 += wt * er[tid];       c1 += wt * er[tid + 256];
        c2 += wt * er[tid + 512]; c3 += wt * er[tid + 768];
    }
    __bf16* arow = a_bf + (size_t)b * NKC;
    arow[NE + tid      ] = (__bf16)(c0 * inv);
    arow[NE + tid + 256] = (__bf16)(c1 * inv);
    arow[NE + tid + 512] = (__bf16)(c2 * inv);
    arow[NE + tid + 768] = (__bf16)(c3 * inv);
    for (int i = tid; i < NE; i += 256) arow[i] = emb_t[b * NE + i];
    for (int i = tid; i < NH; i += 256) arow[NE + NH + i] = h_bf[b * NH + i];
}

// ------------------------------ LSTM cell ----------------------------------
__global__ void lstm_kernel(const float* __restrict__ gates,
                            const float* __restrict__ b_ih, const float* __restrict__ b_hh,
                            float* __restrict__ h, float* __restrict__ c,
                            __bf16* __restrict__ h_bf)
{
    int gid = blockIdx.x * blockDim.x + threadIdx.x;
    if (gid >= NB * NH) return;
    int b = gid >> 10, k = gid & (NH - 1);
    const float* gr = gates + (size_t)b * NG;
    float ig = gr[k]          + b_ih[k]          + b_hh[k];
    float fg = gr[NH + k]     + b_ih[NH + k]     + b_hh[NH + k];
    float gg = gr[2 * NH + k] + b_ih[2 * NH + k] + b_hh[2 * NH + k];
    float og = gr[3 * NH + k] + b_ih[3 * NH + k] + b_hh[3 * NH + k];
    float cn = sigmoidf(fg) * c[gid] + sigmoidf(ig) * tanhf(gg);
    float hn = sigmoidf(og) * tanhf(cn);
    c[gid] = cn;  h[gid] = hn;  h_bf[gid] = (__bf16)hn;
}

// -------------------------- log-softmax over vocab -------------------------
__global__ __launch_bounds__(256)
void logsoftmax_kernel(const float* __restrict__ logits, const float* __restrict__ fc_b,
                       float* __restrict__ out, int t)
{
    __shared__ float s_red[256];
    __shared__ float s_max, s_sum;
    int b = blockIdx.x, tid = threadIdx.x;
    const float* row = logits + (size_t)b * NV;

    float m = -3.4e38f;
    for (int v = tid; v < NV; v += 256) m = fmaxf(m, row[v] + fc_b[v]);
    s_red[tid] = m;  __syncthreads();
    for (int s = 128; s; s >>= 1) { if (tid < s) s_red[tid] = fmaxf(s_red[tid], s_red[tid + s]); __syncthreads(); }
    if (tid == 0) s_max = s_red[0];
    __syncthreads();

    float mm = s_max, sum = 0.f;
    for (int v = tid; v < NV; v += 256) sum += expf(row[v] + fc_b[v] - mm);
    s_red[tid] = sum;  __syncthreads();
    for (int s = 128; s; s >>= 1) { if (tid < s) s_red[tid] += s_red[tid + s]; __syncthreads(); }
    if (tid == 0) s_sum = s_red[0];
    __syncthreads();

    float lse = mm + logf(s_sum);
    float* orow = out + ((size_t)b * NT + t) * NV;
    for (int v = tid; v < NV; v += 256) orow[v] = row[v] + fc_b[v] - lse;
}

// ---------------------------------------------------------------------------
extern "C" void kernel_launch(void* const* d_in, const int* in_sizes, int n_in,
                              void* d_out, int out_size, void* d_ws, size_t ws_size,
                              hipStream_t stream)
{
    (void)in_sizes; (void)n_in; (void)out_size; (void)ws_size;
    const float* enc_out = (const float*)d_in[0];   // [B,TENC,H]
    const float* enc_h   = (const float*)d_in[1];   // [1,B,H]
    const float* enc_c   = (const float*)d_in[2];   // [1,B,H]
    const int*   target  = (const int*)  d_in[3];   // [B,T]
    const float* emb_tab = (const float*)d_in[4];   // [V,E]
    const float* Wa      = (const float*)d_in[5];   // [H,H]
    const float* Ua      = (const float*)d_in[6];   // [H,H]
    const float* Va_w    = (const float*)d_in[7];   // [1,H]
    const float* Va_b    = (const float*)d_in[8];   // [1]
    const float* W_ih    = (const float*)d_in[9];   // [4H, E+H]
    const float* W_hh    = (const float*)d_in[10];  // [4H, H]
    const float* b_ih    = (const float*)d_in[11];  // [4H]
    const float* b_hh    = (const float*)d_in[12];  // [4H]
    const float* fc_w    = (const float*)d_in[13];  // [V,H]
    const float* fc_b    = (const float*)d_in[14];  // [V]
    float* out = (float*)d_out;                     // [B,T,V]

    // workspace carve-out (256B aligned), ~119 MB total
    char* p = (char*)d_ws;
    auto take = [&](size_t bytes) { void* r = p; p += (bytes + 255) & ~(size_t)255; return r; };
    float*  h       = (float*) take((size_t)NB * NH * 4);
    float*  c       = (float*) take((size_t)NB * NH * 4);
    __bf16* h_bf    = (__bf16*)take((size_t)NB * NH * 2);
    __bf16* a_bf    = (__bf16*)take((size_t)NB * NKC * 2);
    float*  qbuf    = (float*) take((size_t)NB * NH * 4);
    float*  gates   = (float*) take((size_t)NB * NG * 4);
    float*  logits  = (float*) take((size_t)NB * NV * 4);
    float*  encUa   = (float*) take((size_t)NB * NTE * NH * 4);
    __bf16* enc_bf  = (__bf16*)take((size_t)NB * NTE * NH * 2);
    __bf16* emb_bf  = (__bf16*)take((size_t)NT * NB * NE * 2);
    __bf16* Wa_bf   = (__bf16*)take((size_t)NH * NH * 2);
    __bf16* Ua_bf   = (__bf16*)take((size_t)NH * NH * 2);
    __bf16* wcat_bf = (__bf16*)take((size_t)NG * NKC * 2);
    __bf16* fcw_bf  = (__bf16*)take((size_t)NV * NH * 2);

    auto launch_gemm = [&](float* C, const __bf16* A, const __bf16* W,
                           int M, int N, int K, int lda, int ldw, int ldc) {
        int mp = M / 32, nt = N / 16;                 // wave = 32x16 C slab
        int blocks = (mp * nt * 32 + 255) / 256;
        wmma_gemm_nt<<<blocks, 256, 0, stream>>>(C, A, W, mp, nt, K, lda, ldw, ldc);
    };

    // ---- one-time precompute (re-done every call; deterministic) ----
    cvt_f32_bf16<<<2048, 256, 0, stream>>>(fcw_bf, fc_w, NV * NH);
    cvt_f32_bf16<<<256,  256, 0, stream>>>(Wa_bf,  Wa,   NH * NH);
    cvt_f32_bf16<<<256,  256, 0, stream>>>(Ua_bf,  Ua,   NH * NH);
    cvt_f32_bf16<<<512,  256, 0, stream>>>(enc_bf, enc_out, NB * NTE * NH);
    build_wcat<<<1024, 256, 0, stream>>>(wcat_bf, W_ih, W_hh);
    build_emb<<<256, 256, 0, stream>>>(emb_bf, target, emb_tab);
    init_state<<<(NB * NH + 255) / 256, 256, 0, stream>>>(h, c, h_bf, enc_h, enc_c);
    // enc_Ua[b,t,k] = sum_h enc[b,t,h] * Ua[k,h]   (M=4096, N=1024, K=1024)
    launch_gemm(encUa, enc_bf, Ua_bf, NB * NTE, NH, NH, NH, NH, NH);

    // ---- 64 sequential decode steps ----
    for (int t = 0; t < NT; ++t) {
        launch_gemm(qbuf, h_bf, Wa_bf, NB, NH, NH, NH, NH, NH);         // q = h Wa^T
        attn_kernel<<<NB, 256, 0, stream>>>(qbuf, encUa, enc_out, Va_w, Va_b,
                                            emb_bf + (size_t)t * NB * NE, h_bf, a_bf);
        launch_gemm(gates, a_bf, wcat_bf, NB, NG, NKC, NKC, NKC, NG);   // gates GEMM
        lstm_kernel<<<(NB * NH + 255) / 256, 256, 0, stream>>>(gates, b_ih, b_hh, h, c, h_bf);
        launch_gemm(logits, h_bf, fcw_bf, NB, NV, NH, NH, NH, NV);      // vocab GEMM
        logsoftmax_kernel<<<NB, 256, 0, stream>>>(logits, fc_b, out, t);
    }
}
